// GELU_82214263980556
// MI455X (gfx1250) — compile-verified, matrix operand order checked by reading, not run on hardware
//
#include <hip/hip_runtime.h>

// ---------------------------------------------------------------------------
// Q8Linear (int8 weight, fp32 activation) + fused tanh-GELU for MI455X/gfx1250
//   C[M,N] = gelu( (A[M,K] . Wq[N,K]^T) * scale[N] + bias[N] )
//   M=8192, K=4096, N=16384
//
// Fast path (needs ~201MB workspace):
//   1) convert A fp32->f16 and Wq int32->f16 into d_ws (one streaming pass)
//   2) GEMM with v_wmma_f32_16x16x32_f16, BK=64 tiles, double-buffered LDS
//      filled by GLOBAL_LOAD_ASYNC_TO_LDS_B128 (ASYNCcnt), 1 barrier/iter.
// Fallback: fused kernel with VGPR-staged conversion.
// ---------------------------------------------------------------------------

typedef __attribute__((ext_vector_type(4)))  _Float16 v4h;
typedef __attribute__((ext_vector_type(8)))  _Float16 v8h;
typedef __attribute__((ext_vector_type(16))) _Float16 v16h;
typedef __attribute__((ext_vector_type(8)))  float    v8f;

constexpr int Mdim = 8192;
constexpr int Ndim = 16384;
constexpr int Kdim = 4096;

constexpr int BM = 128;
constexpr int BN = 128;
constexpr int THREADS = 256;     // 8 wave32

// ---------------- CDNA5 async helpers ----------------

__device__ __forceinline__ uint32_t lds_byte_off(const void* p) {
  // HW maps flat->LDS by truncating to addr[31:0]; low 32 bits of a generic
  // pointer to a __shared__ object are the LDS byte offset.
  return (uint32_t)(uintptr_t)p;
}

__device__ __forceinline__ void async_copy_b128(uint32_t lds_off, const void* gaddr) {
  asm volatile("global_load_async_to_lds_b128 %0, %1, off"
               :: "v"(lds_off), "v"((unsigned long long)(uintptr_t)gaddr)
               : "memory");
}

__device__ __forceinline__ void wait_async0() {
#if __has_builtin(__builtin_amdgcn_s_wait_asynccnt)
  __builtin_amdgcn_s_wait_asynccnt(0);
#else
  asm volatile("s_wait_asynccnt 0x0" ::: "memory");
#endif
}

// ---------------- conversion pass (fast path) ----------------

__global__ __launch_bounds__(256)
void convert_a_kernel(const float* __restrict__ A, _Float16* __restrict__ out,
                      long long n8) {
  long long i = (long long)blockIdx.x * 256 + threadIdx.x;   // 8 elems each
  if (i >= n8) return;
  const float4* src = (const float4*)A;
  float4 a = src[2 * i], b = src[2 * i + 1];
  v8h h = { (_Float16)a.x, (_Float16)a.y, (_Float16)a.z, (_Float16)a.w,
            (_Float16)b.x, (_Float16)b.y, (_Float16)b.z, (_Float16)b.w };
  *(v8h*)(out + 8 * i) = h;
}

__global__ __launch_bounds__(256)
void convert_w_kernel(const int* __restrict__ W, _Float16* __restrict__ out,
                      long long n8) {
  long long i = (long long)blockIdx.x * 256 + threadIdx.x;
  if (i >= n8) return;
  const int4* src = (const int4*)W;
  int4 a = src[2 * i], b = src[2 * i + 1];
  v8h h = { (_Float16)a.x, (_Float16)a.y, (_Float16)a.z, (_Float16)a.w,
            (_Float16)b.x, (_Float16)b.y, (_Float16)b.z, (_Float16)b.w };
  *(v8h*)(out + 8 * i) = h;
}

// ---------------- fast GEMM: f16 in, async-to-LDS pipeline ----------------

constexpr int BK2  = 64;                 // two 16x16x32 K-steps per tile
constexpr int LDT2 = BK2 + 8;            // 72 halfs = 144B stride (16B aligned, bank-safe)
constexpr int NKT2 = Kdim / BK2;         // 64 iterations
constexpr uint32_t BUFB = BM * LDT2 * sizeof(_Float16);  // 18432B buffer pitch

__global__ __launch_bounds__(THREADS)
void q8gemm_gelu_async_kernel(const _Float16* __restrict__ Af16,
                              const _Float16* __restrict__ Wf16,
                              const float* __restrict__ scale,
                              const float* __restrict__ bias,
                              float*       __restrict__ out) {
  __shared__ _Float16 As[2][BM * LDT2];   // 2 x 18 KB
  __shared__ _Float16 Ws[2][BN * LDT2];   // 2 x 18 KB (72 KB total of 320 KB/WGP)

  // Group raster: n fastest within groups of 8 n-tiles (W slab stays in L2).
  constexpr int GROUPN = 8;
  constexpr int MT = Mdim / BM;                 // 64
  const int bid    = blockIdx.x;
  const int group  = bid / (GROUPN * MT);
  const int within = bid % (GROUPN * MT);
  const int nt = group * GROUPN + (within % GROUPN);
  const int mt = within / GROUPN;
  const int m0 = mt * BM;
  const int n0 = nt * BN;

  const int tid  = threadIdx.x;
  const int lane = tid & 31;
  const int wid  = tid >> 5;
  const int wm   = (wid >> 2) * 64;   // wave M offset: 0 or 64
  const int wn   = (wid & 3) * 32;    // wave N offset: 0,32,64,96
  const int r16  = lane & 15;
  const int hi   = lane >> 4;

  v8f acc[4][2];
  #pragma unroll
  for (int i = 0; i < 4; ++i)
    #pragma unroll
    for (int j = 0; j < 2; ++j)
      #pragma unroll
      for (int e = 0; e < 8; ++e)
        acc[i][j][e] = 0.0f;

  // Per-thread copy plan: tile is 128 rows x 64 halfs = 1024 16B-chunks;
  // 256 threads x 4 chunks. chunk c: row = c>>3, ck = c&7 (8 halfs each).
  // Only buffer-0 LDS offsets are kept; buffer 1 is base + BUFB (constant),
  // so no dynamically-indexed VGPR arrays (avoids v_movrels in the hot loop).
  const char* pa[4];
  const char* pw[4];
  uint32_t lA0[4], lW0[4];
  #pragma unroll
  for (int i = 0; i < 4; ++i) {
    const int c  = i * THREADS + tid;
    const int rr = c >> 3;
    const int ck = c & 7;
    pa[i] = (const char*)(Af16 + (size_t)(m0 + rr) * Kdim + ck * 8);
    pw[i] = (const char*)(Wf16 + (size_t)(n0 + rr) * Kdim + ck * 8);
    lA0[i] = lds_byte_off(&As[0][rr * LDT2 + ck * 8]);
    lW0[i] = lds_byte_off(&Ws[0][rr * LDT2 + ck * 8]);
  }

  auto issue_async = [&](uint32_t boff) {
    #pragma unroll
    for (int i = 0; i < 4; ++i) {
      async_copy_b128(lA0[i] + boff, pa[i]);
      async_copy_b128(lW0[i] + boff, pw[i]);
      pa[i] += BK2 * sizeof(_Float16);   // advance one K-tile (128B)
      pw[i] += BK2 * sizeof(_Float16);
    }
  };

  // Fragment gather (16-bit WMMA layout): lane l reads two b128 chunks of
  // row l%16 at K-offsets (l/16)*8 and (l/16)*8+16 inside a 32-wide window.
  auto ldfrag = [&](const _Float16* base, int row, int ks) -> v16h {
    const _Float16* p = base + row * LDT2 + ks * 32 + hi * 8;
    v8h c0 = *(const v8h*)p;
    v8h c1 = *(const v8h*)(p + 16);
    return __builtin_shufflevector(c0, c1,
        0,1,2,3,4,5,6,7,8,9,10,11,12,13,14,15);
  };

  auto compute = [&](int b) {
    #pragma unroll
    for (int ks = 0; ks < 2; ++ks) {
      v16h af[4], bf[2];
      #pragma unroll
      for (int i = 0; i < 4; ++i) af[i] = ldfrag(&As[b][0], wm + i * 16 + r16, ks);
      #pragma unroll
      for (int j = 0; j < 2; ++j) bf[j] = ldfrag(&Ws[b][0], wn + j * 16 + r16, ks);
      #pragma unroll
      for (int i = 0; i < 4; ++i)
        #pragma unroll
        for (int j = 0; j < 2; ++j)
          acc[i][j] = __builtin_amdgcn_wmma_f32_16x16x32_f16(
              false, af[i], false, bf[j], (short)0, acc[i][j], false, false);
    }
  };

  // ---- pipeline: issue async for next tile, compute current, 1 barrier ----
  issue_async(0);
  wait_async0();
  __syncthreads();

  // unroll 2: buffer parity (and thus all LDS offsets) become compile-time
  #pragma unroll 2
  for (int kt = 0; kt < NKT2; ++kt) {
    if (kt + 1 < NKT2) issue_async(((kt + 1) & 1) ? BUFB : 0u);  // overlap w/ WMMA
    compute(kt & 1);
    wait_async0();        // own async writes for tile kt+1 complete
    __syncthreads();      // all waves: reads of buf kt done, writes visible
  }

  // ---- epilogue: per-channel dequant + bias + tanh-GELU ----
  float sc[2], bs[2];
  int   nc[2];
  #pragma unroll
  for (int j = 0; j < 2; ++j) {
    nc[j] = n0 + wn + j * 16 + r16;
    sc[j] = scale[nc[j]];
    bs[j] = bias[nc[j]];
  }
  #pragma unroll
  for (int i = 0; i < 4; ++i)
    #pragma unroll
    for (int j = 0; j < 2; ++j)
      #pragma unroll
      for (int v = 0; v < 8; ++v) {
        const int mrow = m0 + wm + i * 16 + hi * 8 + v;
        float y = acc[i][j][v] * sc[j] + bs[j];
        float u = 0.7978845608028654f * (y + 0.044715f * y * y * y);
        out[(size_t)mrow * Ndim + nc[j]] = 0.5f * y * (1.0f + tanhf(u));
      }
}

// ---------------- fallback: fused conversion GEMM ----------------

constexpr int BK1  = 32;
constexpr int LDT1 = BK1 + 8;            // 40 halfs = 80B stride
constexpr int NKT1 = Kdim / BK1;         // 128

__global__ __launch_bounds__(THREADS)
void q8gemm_gelu_fused_kernel(const float* __restrict__ A,
                              const int*   __restrict__ Wq,
                              const float* __restrict__ scale,
                              const float* __restrict__ bias,
                              float*       __restrict__ out) {
  __shared__ _Float16 As[2][BM * LDT1];
  __shared__ _Float16 Ws[2][BN * LDT1];

  constexpr int GROUPN = 8;
  constexpr int MT = Mdim / BM;
  const int bid    = blockIdx.x;
  const int group  = bid / (GROUPN * MT);
  const int within = bid % (GROUPN * MT);
  const int nt = group * GROUPN + (within % GROUPN);
  const int mt = within / GROUPN;
  const int m0 = mt * BM;
  const int n0 = nt * BN;

  const int tid  = threadIdx.x;
  const int lane = tid & 31;
  const int wid  = tid >> 5;
  const int wm   = (wid >> 2) * 64;
  const int wn   = (wid & 3) * 32;
  const int r16  = lane & 15;
  const int hi   = lane >> 4;

  v8f acc[4][2];
  #pragma unroll
  for (int i = 0; i < 4; ++i)
    #pragma unroll
    for (int j = 0; j < 2; ++j)
      #pragma unroll
      for (int e = 0; e < 8; ++e)
        acc[i][j][e] = 0.0f;

  float4 areg[4];
  int4   wreg[4];

  auto gload = [&](int kt) {
    const int kk = kt * BK1;
    #pragma unroll
    for (int i = 0; i < 4; ++i) {
      const int gi = i * THREADS + tid;
      const int rr = gi >> 3;
      const int c4 = (gi & 7) * 4;
      const float* ap = A  + (size_t)(m0 + rr) * Kdim + kk + c4;
      const int*   wp = Wq + (size_t)(n0 + rr) * Kdim + kk + c4;
      areg[i] = *(const float4*)ap;
      wreg[i] = *(const int4*)wp;
      if (i == 0 && kt + 1 < NKT1) {
        __builtin_prefetch(ap + BK1, 0, 1);
        __builtin_prefetch(wp + BK1, 0, 1);
      }
    }
  };

  auto lstore = [&](int b) {
    #pragma unroll
    for (int i = 0; i < 4; ++i) {
      const int gi = i * THREADS + tid;
      const int rr = gi >> 3;
      const int c4 = (gi & 7) * 4;
      v4h av = { (_Float16)areg[i].x, (_Float16)areg[i].y,
                 (_Float16)areg[i].z, (_Float16)areg[i].w };
      v4h wv = { (_Float16)wreg[i].x, (_Float16)wreg[i].y,
                 (_Float16)wreg[i].z, (_Float16)wreg[i].w };
      *(v4h*)(&As[b][rr * LDT1 + c4]) = av;
      *(v4h*)(&Ws[b][rr * LDT1 + c4]) = wv;
    }
  };

  auto ldfrag = [&](const _Float16* base, int row) -> v16h {
    const _Float16* p = base + row * LDT1 + hi * 8;
    v8h c0 = *(const v8h*)p;
    v8h c1 = *(const v8h*)(p + 16);
    return __builtin_shufflevector(c0, c1,
        0,1,2,3,4,5,6,7,8,9,10,11,12,13,14,15);
  };

  auto compute = [&](int b) {
    v16h af[4], bf[2];
    #pragma unroll
    for (int i = 0; i < 4; ++i) af[i] = ldfrag(&As[b][0], wm + i * 16 + r16);
    #pragma unroll
    for (int j = 0; j < 2; ++j) bf[j] = ldfrag(&Ws[b][0], wn + j * 16 + r16);
    #pragma unroll
    for (int i = 0; i < 4; ++i)
      #pragma unroll
      for (int j = 0; j < 2; ++j)
        acc[i][j] = __builtin_amdgcn_wmma_f32_16x16x32_f16(
            false, af[i], false, bf[j], (short)0, acc[i][j], false, false);
  };

  gload(0);
  lstore(0);
  __syncthreads();

  for (int kt = 0; kt < NKT1; ++kt) {
    if (kt + 1 < NKT1) gload(kt + 1);
    compute(kt & 1);
    __syncthreads();
    if (kt + 1 < NKT1) {
      lstore((kt + 1) & 1);
      __syncthreads();
    }
  }

  float sc[2], bs[2];
  int   nc[2];
  #pragma unroll
  for (int j = 0; j < 2; ++j) {
    nc[j] = n0 + wn + j * 16 + r16;
    sc[j] = scale[nc[j]];
    bs[j] = bias[nc[j]];
  }
  #pragma unroll
  for (int i = 0; i < 4; ++i)
    #pragma unroll
    for (int j = 0; j < 2; ++j)
      #pragma unroll
      for (int v = 0; v < 8; ++v) {
        const int mrow = m0 + wm + i * 16 + hi * 8 + v;
        float y = acc[i][j][v] * sc[j] + bs[j];
        float u = 0.7978845608028654f * (y + 0.044715f * y * y * y);
        out[(size_t)mrow * Ndim + nc[j]] = 0.5f * y * (1.0f + tanhf(u));
      }
}

// ---------------- launch ----------------

extern "C" void kernel_launch(void* const* d_in, const int* in_sizes, int n_in,
                              void* d_out, int out_size, void* d_ws, size_t ws_size,
                              hipStream_t stream) {
  const float* A     = (const float*)d_in[0];   // [8192,4096] f32
  const int*   Wq    = (const int*)  d_in[1];   // [16384,4096] int32
  const float* scale = (const float*)d_in[2];   // [16384] f32
  const float* bias  = (const float*)d_in[3];   // [16384] f32
  float*       out   = (float*)d_out;           // [8192,16384] f32

  constexpr long long A_elems = (long long)Mdim * Kdim;   // 33,554,432
  constexpr long long W_elems = (long long)Ndim * Kdim;   // 67,108,864
  const size_t need = (size_t)(A_elems + W_elems) * sizeof(_Float16); // ~201MB
  const int grid = (Mdim / BM) * (Ndim / BN);             // 8192 blocks

  if (d_ws != nullptr && ws_size >= need) {
    _Float16* Af16 = (_Float16*)d_ws;
    _Float16* Wf16 = Af16 + A_elems;
    convert_a_kernel<<<(int)(A_elems / 8 / 256), 256, 0, stream>>>(A, Af16, A_elems / 8);
    convert_w_kernel<<<(int)(W_elems / 8 / 256), 256, 0, stream>>>(Wq, Wf16, W_elems / 8);
    q8gemm_gelu_async_kernel<<<grid, THREADS, 0, stream>>>(Af16, Wf16, scale, bias, out);
  } else {
    q8gemm_gelu_fused_kernel<<<grid, THREADS, 0, stream>>>(A, Wq, scale, bias, out);
  }
}
